// MultilayerGRU_67018669687003
// MI455X (gfx1250) — matrix operand order repaired and checked by reading, not compiled, not run on hardware
//
#include <hip/hip_runtime.h>
#include <hip/hip_bf16.h>

// ---------------- types ----------------
typedef __attribute__((ext_vector_type(16))) __bf16 v16bf;
typedef __attribute__((ext_vector_type(8)))  __bf16 v8bf;
typedef __attribute__((ext_vector_type(4)))  __bf16 v4bf;
typedef __attribute__((ext_vector_type(8)))  float  v8f;
typedef __attribute__((ext_vector_type(4)))  float  v4f;

#define Hd 1024
#define Bd 64
#define Sd 512
#define Od 512
#define Ld 3

// f32 -> bf16 round-to-nearest-even
__device__ __forceinline__ __bf16 f2bf(float f) {
  union { float f; unsigned u; } v; v.f = f;
  unsigned r = v.u + 0x7FFFu + ((v.u >> 16) & 1u);
  unsigned short s = (unsigned short)(r >> 16);
  __bf16 o; __builtin_memcpy(&o, &s, sizeof(o));
  return o;
}

// ---------------- WMMA fragment loaders (ISA 7.12.2 layouts, wave32) ----------------
// A-matrix 16x32 bf16: lanes 0-15 row=lane, K=[k0..k0+7]+[k0+16..k0+23];
// lanes 16-31 row=lane-16, K=[k0+8..k0+15]+[k0+24..k0+31]. Two b128 loads.
__device__ __forceinline__ v16bf load_a_bf16(const __bf16* __restrict__ p, long ld,
                                             int rowBase, int k0) {
  int lane = threadIdx.x & 31;
  int laneLo = lane & 15;
  int hi = lane >> 4;
  const __bf16* r = p + (long)(rowBase + laneLo) * ld + k0 + hi * 8;
  v8bf lo = *(const v8bf*)r;
  v8bf hv = *(const v8bf*)(r + 16);
  return __builtin_shufflevector(lo, hv, 0,1,2,3,4,5,6,7,8,9,10,11,12,13,14,15);
}

// B-matrix 32x16 (KxN), B[k,n] = W[n,k] (row-major W => x@W^T):
// lanes 0-15 col=lane K=[k0..k0+7]+[k0+16..k0+23]; lanes 16-31 col=lane-16 K=[k0+8..k0+15]+[k0+24..k0+31].
__device__ __forceinline__ v16bf load_b_w(const __bf16* __restrict__ w, int ld,
                                          int colBase, int k0) {
  int lane = threadIdx.x & 31;
  int laneLo = lane & 15;
  int hi = lane >> 4;
  const __bf16* r = w + (size_t)(colBase + laneLo) * ld + k0 + hi * 8;
  v8bf lo = *(const v8bf*)r;
  v8bf hv = *(const v8bf*)(r + 16);
  return __builtin_shufflevector(lo, hv, 0,1,2,3,4,5,6,7,8,9,10,11,12,13,14,15);
}

__device__ __forceinline__ v8f wmma_bf16(v16bf a, v16bf b, v8f c) {
  return __builtin_amdgcn_wmma_f32_16x16x32_bf16(false, a, false, b, (short)0, c,
                                                 false, false);
}

// ---------------- pre/post passes ----------------
__global__ void __launch_bounds__(256)
convert_f32_bf16_x4(const float* __restrict__ src, __bf16* __restrict__ dst, int n4) {
  int i = blockIdx.x * blockDim.x + threadIdx.x;
  if (i < n4) {
    v4f v = *(const v4f*)(src + (size_t)i * 4);
    v4bf o;
#pragma unroll
    for (int j = 0; j < 4; ++j) o[j] = f2bf(v[j]);
    *(v4bf*)(dst + (size_t)i * 4) = o;
  }
}

// h0 (B,L,H) -> hst (L,B,H) f32 AND hbf (L,B,H) bf16
__global__ void __launch_bounds__(256)
init_h_kernel(const float* __restrict__ h0, float* __restrict__ hst,
              __bf16* __restrict__ hbf) {
  int i = blockIdx.x * blockDim.x + threadIdx.x;
  if (i < Ld * Bd * Hd) {
    int hh = i % Hd;
    int bb = (i / Hd) % Bd;
    int ll = i / (Bd * Hd);
    float v = h0[(size_t)bb * Ld * Hd + (size_t)ll * Hd + hh];
    hst[i] = v;
    hbf[i] = f2bf(v);
  }
}

// hst (L,B,H) -> out hidden (B,L,H)
__global__ void __launch_bounds__(256)
final_h_kernel(const float* __restrict__ hst, float* __restrict__ out) {
  int i = blockIdx.x * blockDim.x + threadIdx.x;
  if (i < Bd * Ld * Hd) {
    int hh = i % Hd;
    int ll = (i / Hd) % Ld;
    int bb = i / (Ld * Hd);
    out[i] = hst[(size_t)ll * Bd * Hd + (size_t)bb * Hd + hh];
  }
}

// ---------------- GRU cell, phase A ----------------
// block = 128 threads = 4 waves; each wave owns ALL of M (64 rows) x 16 cols.
// grid = (H/64, 2): blockIdx.y == 0 -> z & r gates (x-GEMM + h-GEMM, K=2048 eff),
//                   blockIdx.y == 1 -> x-part of g (K=1024, finishes early).
// Splitting by gate doubles per-step parallelism with bitwise-deterministic sums.
__global__ void __launch_bounds__(128)
gru_gates_kernel(const __bf16* __restrict__ ain, long aLd,   // bf16 input (x slice or prev-layer h)
                 const __bf16* __restrict__ hbf,             // bf16 h state (B,H)
                 const float*  __restrict__ h,               // f32 h state (B,H)
                 const __bf16* __restrict__ Wxz, const __bf16* __restrict__ Wxr,
                 const __bf16* __restrict__ Wxg,
                 const __bf16* __restrict__ Whz, const __bf16* __restrict__ Whr,
                 const float* __restrict__ bhz, const float* __restrict__ bhr,
                 const float* __restrict__ bhg,
                 float* __restrict__ zbuf, float* __restrict__ gxbuf,
                 __bf16* __restrict__ rhbuf) {
  int wave = threadIdx.x >> 5;
  int nCol = blockIdx.x * 64 + wave * 16;  // this wave's 16 output columns
  int lane = threadIdx.x & 31;
  int laneLo = lane & 15;
  int hi = lane >> 4;
  int col = nCol + laneLo;

  if (blockIdx.y == 0) {
    // ---------------- z and r gates ----------------
    v8f accZ[4] = {};
    v8f accR[4] = {};
#pragma unroll 2
    for (int k0 = 0; k0 < Hd; k0 += 32) {
      v16bf a[4];
#pragma unroll
      for (int m = 0; m < 4; ++m) a[m] = load_a_bf16(ain, aLd, m * 16, k0);
      __builtin_prefetch(Wxz + (size_t)nCol * Hd + k0 + 64, 0, 3);
      v16bf bz = load_b_w(Wxz, Hd, nCol, k0);
#pragma unroll
      for (int m = 0; m < 4; ++m) accZ[m] = wmma_bf16(a[m], bz, accZ[m]);
      v16bf br = load_b_w(Wxr, Hd, nCol, k0);
#pragma unroll
      for (int m = 0; m < 4; ++m) accR[m] = wmma_bf16(a[m], br, accR[m]);
    }
#pragma unroll 2
    for (int k0 = 0; k0 < Hd; k0 += 32) {
      v16bf a[4];
#pragma unroll
      for (int m = 0; m < 4; ++m) a[m] = load_a_bf16(hbf, Hd, m * 16, k0);
      __builtin_prefetch(Whz + (size_t)nCol * Hd + k0 + 64, 0, 3);
      v16bf bz = load_b_w(Whz, Hd, nCol, k0);
#pragma unroll
      for (int m = 0; m < 4; ++m) accZ[m] = wmma_bf16(a[m], bz, accZ[m]);
      v16bf br = load_b_w(Whr, Hd, nCol, k0);
#pragma unroll
      for (int m = 0; m < 4; ++m) accR[m] = wmma_bf16(a[m], br, accR[m]);
    }
    float bz = bhz[col], br = bhr[col];
#pragma unroll
    for (int m = 0; m < 4; ++m) {
#pragma unroll
      for (int i = 0; i < 8; ++i) {
        int row = m * 16 + hi * 8 + i;
        size_t idx = (size_t)row * Hd + col;
        float z = 1.f / (1.f + __expf(-(accZ[m][i] + bz)));
        float r = 1.f / (1.f + __expf(-(accR[m][i] + br)));
        zbuf[idx] = z;
        rhbuf[idx] = f2bf(r * h[idx]);
      }
    }
  } else {
    // ---------------- x-part of g gate ----------------
    v8f accG[4] = {};
#pragma unroll 2
    for (int k0 = 0; k0 < Hd; k0 += 32) {
      v16bf a[4];
#pragma unroll
      for (int m = 0; m < 4; ++m) a[m] = load_a_bf16(ain, aLd, m * 16, k0);
      __builtin_prefetch(Wxg + (size_t)nCol * Hd + k0 + 64, 0, 3);
      v16bf bg = load_b_w(Wxg, Hd, nCol, k0);
#pragma unroll
      for (int m = 0; m < 4; ++m) accG[m] = wmma_bf16(a[m], bg, accG[m]);
    }
    float bg = bhg[col];
#pragma unroll
    for (int m = 0; m < 4; ++m) {
#pragma unroll
      for (int i = 0; i < 8; ++i) {
        int row = m * 16 + hi * 8 + i;
        gxbuf[(size_t)row * Hd + col] = accG[m][i] + bg;  // bias folded; tanh in phase B
      }
    }
  }
}

// ---------------- GRU cell, phase B: g GEMM + state update ----------------
// Writes h (f32) and its bf16 image (= next layer's input AND next step's hbf).
__global__ void __launch_bounds__(128)
gru_update_kernel(const __bf16* __restrict__ rh, const __bf16* __restrict__ Whg,
                  const float* __restrict__ gx, const float* __restrict__ zbuf,
                  float* __restrict__ h, __bf16* __restrict__ hbf_out) {
  int wave = threadIdx.x >> 5;
  int nCol = blockIdx.x * 64 + wave * 16;

  v8f acc[4] = {};
#pragma unroll 2
  for (int k0 = 0; k0 < Hd; k0 += 32) {
    v16bf a[4];
#pragma unroll
    for (int m = 0; m < 4; ++m) a[m] = load_a_bf16(rh, Hd, m * 16, k0);
    __builtin_prefetch(Whg + (size_t)nCol * Hd + k0 + 64, 0, 3);
    v16bf b = load_b_w(Whg, Hd, nCol, k0);
#pragma unroll
    for (int m = 0; m < 4; ++m) acc[m] = wmma_bf16(a[m], b, acc[m]);
  }

  int lane = threadIdx.x & 31;
  int laneLo = lane & 15;
  int hi = lane >> 4;
  int col = nCol + laneLo;
#pragma unroll
  for (int m = 0; m < 4; ++m) {
#pragma unroll
    for (int i = 0; i < 8; ++i) {
      int row = m * 16 + hi * 8 + i;
      size_t idx = (size_t)row * Hd + col;
      float g = tanhf(gx[idx] + acc[m][i]);
      float z = zbuf[idx];
      float hn = z * h[idx] + (1.f - z) * g;  // in-place safe: 1 thread/element
      h[idx] = hn;
      hbf_out[idx] = f2bf(hn);
    }
  }
}

// ---------------- output projection: out[:,t,:] = tin @ Wo^T + bo ----------------
__global__ void __launch_bounds__(128)
out_proj_kernel(const __bf16* __restrict__ tin, const __bf16* __restrict__ Wo,
                const float* __restrict__ bo, float* __restrict__ out, int t) {
  int wave = threadIdx.x >> 5;
  int nCol = blockIdx.x * 64 + wave * 16;

  v8f acc[4] = {};
#pragma unroll 2
  for (int k0 = 0; k0 < Hd; k0 += 32) {
    v16bf a[4];
#pragma unroll
    for (int m = 0; m < 4; ++m) a[m] = load_a_bf16(tin, Hd, m * 16, k0);
    v16bf b = load_b_w(Wo, Hd, nCol, k0);
#pragma unroll
    for (int m = 0; m < 4; ++m) acc[m] = wmma_bf16(a[m], b, acc[m]);
  }

  int lane = threadIdx.x & 31;
  int laneLo = lane & 15;
  int hi = lane >> 4;
  int col = nCol + laneLo;
  float b = bo[col];
#pragma unroll
  for (int m = 0; m < 4; ++m) {
#pragma unroll
    for (int i = 0; i < 8; ++i) {
      int row = m * 16 + hi * 8 + i;
      out[(size_t)row * Sd * Od + (size_t)t * Od + col] = acc[m][i] + b;
    }
  }
}

// ---------------- host launch ----------------
extern "C" void kernel_launch(void* const* d_in, const int* in_sizes, int n_in,
                              void* d_out, int out_size, void* d_ws, size_t ws_size,
                              hipStream_t stream) {
  const float* x   = (const float*)d_in[0];
  const float* h0  = (const float*)d_in[1];
  const float* Wxz = (const float*)d_in[2];
  const float* Wxr = (const float*)d_in[3];
  const float* Wxg = (const float*)d_in[4];
  const float* Whz = (const float*)d_in[5];
  const float* Whr = (const float*)d_in[6];
  const float* Whg = (const float*)d_in[7];
  const float* bhz = (const float*)d_in[8];
  const float* bhr = (const float*)d_in[9];
  const float* bhg = (const float*)d_in[10];
  const float* Wo  = (const float*)d_in[11];
  const float* bo  = (const float*)d_in[12];
  float* out = (float*)d_out;
  (void)in_sizes; (void)n_in; (void)out_size; (void)ws_size;

  char* ws = (char*)d_ws;
  size_t off = 0;
  auto alloc = [&](size_t bytes) -> char* {
    off = (off + 255) & ~(size_t)255;
    char* p = ws + off;
    off += bytes;
    return p;
  };

  const size_t WHH = (size_t)Ld * Hd * Hd;
  __bf16* Wxz_b = (__bf16*)alloc(WHH * 2);
  __bf16* Wxr_b = (__bf16*)alloc(WHH * 2);
  __bf16* Wxg_b = (__bf16*)alloc(WHH * 2);
  __bf16* Whz_b = (__bf16*)alloc(WHH * 2);
  __bf16* Whr_b = (__bf16*)alloc(WHH * 2);
  __bf16* Whg_b = (__bf16*)alloc(WHH * 2);
  __bf16* Wo_b  = (__bf16*)alloc((size_t)Od * Hd * 2);
  __bf16* x_b   = (__bf16*)alloc((size_t)Bd * Sd * Hd * 2);   // bf16 image of x
  float*  hst   = (float*)alloc((size_t)Ld * Bd * Hd * 4);    // f32 h (L,B,H)
  __bf16* hbf   = (__bf16*)alloc((size_t)Ld * Bd * Hd * 2);   // bf16 h (L,B,H)
  float*  zbuf  = (float*)alloc((size_t)Bd * Hd * 4);
  float*  gxbuf = (float*)alloc((size_t)Bd * Hd * 4);
  __bf16* rhbuf = (__bf16*)alloc((size_t)Bd * Hd * 2);

  // one-time (per call): weight + x conversion, state init
  const int nW4 = (int)(WHH / 4);
  const int cbW = (nW4 + 255) / 256;
  convert_f32_bf16_x4<<<cbW, 256, 0, stream>>>(Wxz, Wxz_b, nW4);
  convert_f32_bf16_x4<<<cbW, 256, 0, stream>>>(Wxr, Wxr_b, nW4);
  convert_f32_bf16_x4<<<cbW, 256, 0, stream>>>(Wxg, Wxg_b, nW4);
  convert_f32_bf16_x4<<<cbW, 256, 0, stream>>>(Whz, Whz_b, nW4);
  convert_f32_bf16_x4<<<cbW, 256, 0, stream>>>(Whr, Whr_b, nW4);
  convert_f32_bf16_x4<<<cbW, 256, 0, stream>>>(Whg, Whg_b, nW4);
  const int nWo4 = Od * Hd / 4;
  convert_f32_bf16_x4<<<(nWo4 + 255) / 256, 256, 0, stream>>>(Wo, Wo_b, nWo4);
  const int nX4 = (int)((size_t)Bd * Sd * Hd / 4);
  convert_f32_bf16_x4<<<(nX4 + 255) / 256, 256, 0, stream>>>(x, x_b, nX4);
  init_h_kernel<<<(Ld * Bd * Hd + 255) / 256, 256, 0, stream>>>(h0, hst, hbf);

  const dim3 gatesGrid(Hd / 64, 2);
  for (int t = 0; t < Sd; ++t) {
    const __bf16* ain = x_b + (size_t)t * Hd;  // layer 0: row b at stride S*H
    long aLd = (long)Sd * Hd;
    for (int l = 0; l < Ld; ++l) {
      const size_t wo = (size_t)l * Hd * Hd;
      float*  h_l  = hst + (size_t)l * Bd * Hd;
      __bf16* hb_l = hbf + (size_t)l * Bd * Hd;
      gru_gates_kernel<<<gatesGrid, 128, 0, stream>>>(
          ain, aLd, hb_l, h_l,
          Wxz_b + wo, Wxr_b + wo, Wxg_b + wo, Whz_b + wo, Whr_b + wo,
          bhz + (size_t)l * Hd, bhr + (size_t)l * Hd, bhg + (size_t)l * Hd,
          zbuf, gxbuf, rhbuf);
      gru_update_kernel<<<Hd / 64, 128, 0, stream>>>(
          rhbuf, Whg_b + wo, gxbuf, zbuf, h_l, hb_l);
      ain = hb_l;   // next layer's input = this layer's new h (bf16)
      aLd = Hd;
    }
    out_proj_kernel<<<Od / 64, 128, 0, stream>>>(hbf + (size_t)(Ld - 1) * Bd * Hd,
                                                 Wo_b, bo, out, t);
  }
  final_h_kernel<<<(Bd * Ld * Hd + 255) / 256, 256, 0, stream>>>(
      hst, out + (size_t)Bd * Sd * Od);
}
